// SetAbstraction_28587302322365
// MI455X (gfx1250) — matrix operand order repaired and checked by reading, not compiled, not run on hardware
//
#include <hip/hip_runtime.h>
#include <hip/hip_bf16.h>

// ---------------------------------------------------------------------------
// PointNet++ SetAbstraction for MI455X (gfx1250, wave32, WMMA)
//   B=16, N=4096, D=64, NPOINT=1024, RADIUS=0.2, NSAMPLE=32, MLP 67->64->64->128
// ---------------------------------------------------------------------------

typedef __attribute__((ext_vector_type(16))) _Float16 v16h;
typedef __attribute__((ext_vector_type(8)))  _Float16 v8h;
typedef __attribute__((ext_vector_type(8)))  float    v8f;

#define NB    16
#define NN    4096
#define NS    1024
#define KNBR  32
#define R2    0.04f
#define C0    64          // input feature channels
#define CP1   96          // padded input channels (3 + 64 -> 96)
#define H1    64
#define H2    64
#define H3    128

// ---------------------------------------------------------------------------
// Kernel 0: fold BN into weights, cast to f16, compute effective biases.
//   y = relu( (s*W) x + (s*(b-mean)+beta) ),  s = gamma*rsqrt(var+eps)
// ---------------------------------------------------------------------------
__global__ void prep_weights(
    const float* __restrict__ w1, const float* __restrict__ b1,
    const float* __restrict__ g1, const float* __restrict__ be1,
    const float* __restrict__ m1, const float* __restrict__ v1,
    const float* __restrict__ w2, const float* __restrict__ b2,
    const float* __restrict__ g2, const float* __restrict__ be2,
    const float* __restrict__ m2, const float* __restrict__ v2,
    const float* __restrict__ w3, const float* __restrict__ b3,
    const float* __restrict__ g3, const float* __restrict__ be3,
    const float* __restrict__ m3, const float* __restrict__ v3,
    _Float16* __restrict__ W1h, _Float16* __restrict__ W2h, _Float16* __restrict__ W3h,
    float* __restrict__ B1f, float* __restrict__ B2f, float* __restrict__ B3f)
{
    const int tid = threadIdx.x + blockIdx.x * blockDim.x;
    const int stride = blockDim.x * gridDim.x;
    // layer 1: [64][67] -> padded [64][96]
    for (int i = tid; i < H1 * CP1; i += stride) {
        int o = i / CP1, c = i % CP1;
        float s = g1[o] * rsqrtf(v1[o] + 1e-5f);
        W1h[i] = (_Float16)((c < 67) ? w1[o * 67 + c] * s : 0.0f);
    }
    for (int o = tid; o < H1; o += stride) {
        float s = g1[o] * rsqrtf(v1[o] + 1e-5f);
        B1f[o] = (b1[o] - m1[o]) * s + be1[o];
    }
    // layer 2: [64][64]
    for (int i = tid; i < H2 * H1; i += stride) {
        int o = i / H1;
        float s = g2[o] * rsqrtf(v2[o] + 1e-5f);
        W2h[i] = (_Float16)(w2[i] * s);
    }
    for (int o = tid; o < H2; o += stride) {
        float s = g2[o] * rsqrtf(v2[o] + 1e-5f);
        B2f[o] = (b2[o] - m2[o]) * s + be2[o];
    }
    // layer 3: [128][64]
    for (int i = tid; i < H3 * H2; i += stride) {
        int o = i / H2;
        float s = g3[o] * rsqrtf(v3[o] + 1e-5f);
        W3h[i] = (_Float16)(w3[i] * s);
    }
    for (int o = tid; o < H3; o += stride) {
        float s = g3[o] * rsqrtf(v3[o] + 1e-5f);
        B3f[o] = (b3[o] - m3[o]) * s + be3[o];
    }
}

// ---------------------------------------------------------------------------
// Kernel 1: farthest point sampling. One block per batch; dist[] in LDS.
// Writes new_xyz output (B,3,NS) directly (first output region).
// ---------------------------------------------------------------------------
__global__ __launch_bounds__(256) void fps_kernel(
    const float* __restrict__ xyz,   // (B,3,N)
    float* __restrict__ new_xyz)     // (B,3,NS)
{
    __shared__ float dist[NN];
    __shared__ float rv[8];
    __shared__ int   ri[8];
    __shared__ int   sfar;

    const int b    = blockIdx.x;
    const int tid  = threadIdx.x;
    const int lane = tid & 31;
    const int wv   = tid >> 5;
    const float* xb = xyz + (size_t)b * 3 * NN;

    for (int n = tid; n < NN; n += 256) dist[n] = 1e10f;
    if (tid == 0) sfar = 0;
    __syncthreads();

    for (int p = 0; p < NS; ++p) {
        const int far = sfar;
        const float cx = xb[far];
        const float cy = xb[NN + far];
        const float cz = xb[2 * NN + far];
        if (tid == 0) {
            float* o = new_xyz + (size_t)b * 3 * NS;
            o[p] = cx; o[NS + p] = cy; o[2 * NS + p] = cz;
        }
        float bv = -1.0f; int bi = 0;
        #pragma unroll
        for (int t = 0; t < NN / 256; ++t) {
            const int n = tid + t * 256;
            const float dx = xb[n] - cx;
            const float dy = xb[NN + n] - cy;
            const float dz = xb[2 * NN + n] - cz;
            const float d2 = dx * dx + dy * dy + dz * dz;
            const float nd = fminf(dist[n], d2);
            dist[n] = nd;
            if (nd > bv) { bv = nd; bi = n; }
        }
        // wave32 reduce (prefer lower index on ties, matching jnp.argmax)
        #pragma unroll
        for (int off = 16; off > 0; off >>= 1) {
            float ov = __shfl_xor(bv, off);
            int   oi = __shfl_xor(bi, off);
            if (ov > bv || (ov == bv && oi < bi)) { bv = ov; bi = oi; }
        }
        if (lane == 0) { rv[wv] = bv; ri[wv] = bi; }
        __syncthreads();
        if (tid == 0) {
            float mv = rv[0]; int mi = ri[0];
            #pragma unroll
            for (int w = 1; w < 8; ++w)
                if (rv[w] > mv || (rv[w] == mv && ri[w] < mi)) { mv = rv[w]; mi = ri[w]; }
            sfar = mi;
        }
        __syncthreads();
    }
}

// ---------------------------------------------------------------------------
// Kernel 2: ball query. One wave per centroid; first 32 indices with d2<=r2
// in ascending index order (exact match to reference sort-then-take semantics).
// ---------------------------------------------------------------------------
__global__ __launch_bounds__(256) void ball_query_kernel(
    const float* __restrict__ xyz,       // (B,3,N)
    const float* __restrict__ new_xyz,   // (B,3,NS)
    int* __restrict__ idx_out)           // (B*NS, 32)
{
    __shared__ int ibuf[8][KNBR];
    const int lane = threadIdx.x & 31;
    const int wv   = threadIdx.x >> 5;
    const int sg   = blockIdx.x * 8 + wv;     // global centroid id
    const int b    = sg >> 10;
    const int s    = sg & (NS - 1);
    const float* xb = xyz + (size_t)b * 3 * NN;
    const float* nx = new_xyz + (size_t)b * 3 * NS;
    const float cx = nx[s], cy = nx[NS + s], cz = nx[2 * NS + s];

    int cnt = 0;
    for (int base = 0; base < NN && cnt < KNBR; base += 32) {
        const int n = base + lane;
        const float dx = xb[n] - cx;
        const float dy = xb[NN + n] - cy;
        const float dz = xb[2 * NN + n] - cz;
        const bool in = (dx * dx + dy * dy + dz * dz) <= R2;
        const unsigned mask = __builtin_amdgcn_ballot_w32(in);
        const int pos = cnt + __popc(mask & ((1u << lane) - 1u));
        if (in && pos < KNBR) ibuf[wv][pos] = n;
        cnt += __popc(mask);
    }
    if (cnt == 0 && lane == 0) ibuf[wv][0] = 0;  // defensive; FPS point itself is always in radius
    const int cc = (cnt < KNBR) ? ((cnt < 1) ? 1 : cnt) : KNBR;
    const int v = ibuf[wv][(lane < cc) ? lane : 0];
    idx_out[(size_t)sg * KNBR + lane] = v;
}

// ---------------------------------------------------------------------------
// WMMA fragment loaders (layouts per CDNA5 ISA §7.12.2)
// ---------------------------------------------------------------------------
// A (16x32 f16) from LDS, row-major [row][k], stride in halfs (multiple of 8).
__device__ inline v16h load_a_lds(const _Float16* X, int stride, int mrow0, int kcol0, int lane) {
    const int row = mrow0 + (lane & 15);
    const int kb  = (lane >> 4) ? 8 : 0;
    const _Float16* p = X + row * stride + kcol0 + kb;
    v8h lo = *(const v8h*)(p);        // K = kcol0+kb .. +7
    v8h hi = *(const v8h*)(p + 16);   // K = kcol0+16+kb .. +7
    return __builtin_shufflevector(lo, hi, 0, 1, 2, 3, 4, 5, 6, 7, 8, 9, 10, 11, 12, 13, 14, 15);
}
// B (32x16 f16) from global folded weights W'[O][Cpad]; B[k][n] = W'[n][k].
__device__ inline v16h load_b_glb(const _Float16* __restrict__ W, int cpad, int ncol0, int kcol0, int lane) {
    const int col = ncol0 + (lane & 15);
    const int kb  = (lane >> 4) ? 8 : 0;
    const _Float16* p = W + (size_t)col * cpad + kcol0 + kb;
    v8h lo = *(const v8h*)(p);        // 16B aligned
    v8h hi = *(const v8h*)(p + 16);
    return __builtin_shufflevector(lo, hi, 0, 1, 2, 3, 4, 5, 6, 7, 8, 9, 10, 11, 12, 13, 14, 15);
}

// ---------------------------------------------------------------------------
// Kernel 3: fused gather + MLP (WMMA, f16 in / f32 acc) + ReLU + max-pool.
// One wave per centroid, 4 waves (128 threads) per block.
// ---------------------------------------------------------------------------
__global__ __launch_bounds__(128) void sa_fused_kernel(
    const float* __restrict__ xyz,       // (B,3,N)
    const float* __restrict__ pts,       // (B,64,N)
    const float* __restrict__ new_xyz,   // (B,3,NS)
    const int*   __restrict__ idx,       // (B*NS,32)
    const _Float16* __restrict__ W1h, const float* __restrict__ B1f,
    const _Float16* __restrict__ W2h, const float* __restrict__ B2f,
    const _Float16* __restrict__ W3h, const float* __restrict__ B3f,
    float* __restrict__ out_points)      // (B,128,NS)
{
    __shared__ _Float16 X0[4][KNBR * CP1];   // per-wave 32x96 activations (layers 1 & 3 input)
    __shared__ _Float16 X1[4][KNBR * H1];    // per-wave 32x64 activations (layer 2 input)

    const int lane = threadIdx.x & 31;
    const int wv   = threadIdx.x >> 5;
    const int sg   = blockIdx.x * 4 + wv;
    const int b    = sg >> 10;
    const int s    = sg & (NS - 1);
    _Float16* Xa = X0[wv];
    _Float16* Xb = X1[wv];

    // ---- gather: row = lane (one neighbor per lane) -------------------------
    {
        const int n = idx[(size_t)sg * KNBR + lane];
        const float* xb = xyz + (size_t)b * 3 * NN;
        const float* nx = new_xyz + (size_t)b * 3 * NS;
        _Float16* r = Xa + lane * CP1;
        r[0] = (_Float16)(xb[n]          - nx[s]);
        r[1] = (_Float16)(xb[NN + n]     - nx[NS + s]);
        r[2] = (_Float16)(xb[2 * NN + n] - nx[2 * NS + s]);
        const float* pb = pts + (size_t)b * C0 * NN;
        #pragma unroll
        for (int c = 0; c < C0; ++c) r[3 + c] = (_Float16)pb[(size_t)c * NN + n];
        #pragma unroll
        for (int c = 3 + C0; c < CP1; ++c) r[c] = (_Float16)0.0f;
    }

    const int coll = lane & 15;
    const int hi8  = (lane >> 4) ? 8 : 0;

    // ---- layer 1: 96 -> 64 (X0 -> X1) --------------------------------------
    for (int nt = 0; nt < H1 / 16; ++nt) {
        const float bias = B1f[nt * 16 + coll];
        #pragma unroll
        for (int m = 0; m < 2; ++m) {
            v8f acc;
            #pragma unroll
            for (int j = 0; j < 8; ++j) acc[j] = bias;   // bias via C operand
            #pragma unroll
            for (int kk = 0; kk < CP1 / 32; ++kk) {
                v16h a  = load_a_lds(Xa, CP1, m * 16, kk * 32, lane);
                v16h bm = load_b_glb(W1h, CP1, nt * 16, kk * 32, lane);
                acc = __builtin_amdgcn_wmma_f32_16x16x32_f16(false, a, false, bm, (short)0, acc, false, false);
            }
            const int colg = nt * 16 + coll;
            #pragma unroll
            for (int j = 0; j < 8; ++j)
                Xb[(m * 16 + hi8 + j) * H1 + colg] = (_Float16)fmaxf(acc[j], 0.0f);
        }
    }

    // ---- layer 2: 64 -> 64 (X1 -> X0 cols 0..63) ---------------------------
    for (int nt = 0; nt < H2 / 16; ++nt) {
        const float bias = B2f[nt * 16 + coll];
        #pragma unroll
        for (int m = 0; m < 2; ++m) {
            v8f acc;
            #pragma unroll
            for (int j = 0; j < 8; ++j) acc[j] = bias;
            #pragma unroll
            for (int kk = 0; kk < H1 / 32; ++kk) {
                v16h a  = load_a_lds(Xb, H1, m * 16, kk * 32, lane);
                v16h bm = load_b_glb(W2h, H1, nt * 16, kk * 32, lane);
                acc = __builtin_amdgcn_wmma_f32_16x16x32_f16(false, a, false, bm, (short)0, acc, false, false);
            }
            const int colg = nt * 16 + coll;
            #pragma unroll
            for (int j = 0; j < 8; ++j)
                Xa[(m * 16 + hi8 + j) * CP1 + colg] = (_Float16)fmaxf(acc[j], 0.0f);
        }
    }

    // ---- layer 3: 64 -> 128 + ReLU + max over K=32, straight from acc ------
    float* outp = out_points + (size_t)b * H3 * NS;
    for (int nt = 0; nt < H3 / 16; ++nt) {
        const float bias = B3f[nt * 16 + coll];
        float cmax = 0.0f;                       // ReLU floor == pool identity
        #pragma unroll
        for (int m = 0; m < 2; ++m) {
            v8f acc;
            #pragma unroll
            for (int j = 0; j < 8; ++j) acc[j] = bias;
            #pragma unroll
            for (int kk = 0; kk < H2 / 32; ++kk) {
                v16h a  = load_a_lds(Xa, CP1, m * 16, kk * 32, lane);
                v16h bm = load_b_glb(W3h, H2, nt * 16, kk * 32, lane);
                acc = __builtin_amdgcn_wmma_f32_16x16x32_f16(false, a, false, bm, (short)0, acc, false, false);
            }
            #pragma unroll
            for (int j = 0; j < 8; ++j) cmax = fmaxf(cmax, acc[j]);  // rows m*16+hi8+j
        }
        cmax = fmaxf(cmax, __shfl_xor(cmax, 16));  // merge the two half-wave row groups
        if (lane < 16)
            outp[(size_t)(nt * 16 + lane) * NS + s] = cmax;
    }
}

// ---------------------------------------------------------------------------
// Host launcher
// ---------------------------------------------------------------------------
extern "C" void kernel_launch(void* const* d_in, const int* in_sizes, int n_in,
                              void* d_out, int out_size, void* d_ws, size_t ws_size,
                              hipStream_t stream) {
    (void)in_sizes; (void)n_in; (void)out_size; (void)ws_size;

    const float* xyz = (const float*)d_in[0];   // (16,3,4096)
    const float* pts = (const float*)d_in[1];   // (16,64,4096)
    // params: per layer {w,b,gamma,beta,mean,var} starting at index 2
    const float* w1 = (const float*)d_in[2];  const float* b1 = (const float*)d_in[3];
    const float* g1 = (const float*)d_in[4];  const float* be1 = (const float*)d_in[5];
    const float* m1 = (const float*)d_in[6];  const float* v1 = (const float*)d_in[7];
    const float* w2 = (const float*)d_in[8];  const float* b2 = (const float*)d_in[9];
    const float* g2 = (const float*)d_in[10]; const float* be2 = (const float*)d_in[11];
    const float* m2 = (const float*)d_in[12]; const float* v2 = (const float*)d_in[13];
    const float* w3 = (const float*)d_in[14]; const float* b3 = (const float*)d_in[15];
    const float* g3 = (const float*)d_in[16]; const float* be3 = (const float*)d_in[17];
    const float* m3 = (const float*)d_in[18]; const float* v3 = (const float*)d_in[19];

    float* out      = (float*)d_out;
    float* new_xyz  = out;                       // 16*3*1024
    float* new_pts  = out + (size_t)NB * 3 * NS; // 16*128*1024

    char* ws = (char*)d_ws;
    size_t off = 0;
    int* idx = (int*)(ws + off);            off += (size_t)NB * NS * KNBR * sizeof(int);  // 2 MB
    _Float16* W1h = (_Float16*)(ws + off);  off += (size_t)H1 * CP1 * sizeof(_Float16);
    _Float16* W2h = (_Float16*)(ws + off);  off += (size_t)H2 * H1 * sizeof(_Float16);
    _Float16* W3h = (_Float16*)(ws + off);  off += (size_t)H3 * H2 * sizeof(_Float16);
    float* B1f = (float*)(ws + off);        off += H1 * sizeof(float);
    float* B2f = (float*)(ws + off);        off += H2 * sizeof(float);
    float* B3f = (float*)(ws + off);        off += H3 * sizeof(float);

    prep_weights<<<32, 256, 0, stream>>>(w1, b1, g1, be1, m1, v1,
                                         w2, b2, g2, be2, m2, v2,
                                         w3, b3, g3, be3, m3, v3,
                                         W1h, W2h, W3h, B1f, B2f, B3f);

    fps_kernel<<<NB, 256, 0, stream>>>(xyz, new_xyz);

    ball_query_kernel<<<(NB * NS) / 8, 256, 0, stream>>>(xyz, new_xyz, idx);

    sa_fused_kernel<<<(NB * NS) / 4, 128, 0, stream>>>(xyz, pts, new_xyz, idx,
                                                       W1h, B1f, W2h, B2f, W3h, B3f,
                                                       new_pts);
}
